// SelfAttentionModule_23948737643085
// MI455X (gfx1250) — compile-verified, matrix operand order checked by reading, not run on hardware
//
#include <hip/hip_runtime.h>

// ---------------------------------------------------------------------------
// Self-attention block for MI455X (gfx1250, wave32, WMMA + TDM).
// GEMMs: v_wmma_f32_16x16x32_bf16, A/B tiles staged to LDS by the Tensor Data
// Mover (double-buffered, TENSORcnt-synchronized), bf16 intermediates.
// ---------------------------------------------------------------------------

#define SLEN  97
#define BATCH 256
#define WDIM  1024
#define NHEAD 16
#define HDIM  64
#define NTOK  (SLEN * BATCH)           // 24832 rows = 194 * 128 exactly

static const size_t TOKW = (size_t)NTOK * WDIM;   // activation elements
static const size_t WW   = (size_t)WDIM * WDIM;   // weight elements

typedef __attribute__((ext_vector_type(16))) __bf16         v16bf;
typedef __attribute__((ext_vector_type(8)))  float          v8f;
typedef __attribute__((ext_vector_type(4)))  unsigned int   u32x4;
typedef __attribute__((ext_vector_type(4)))  float          f32x4;
typedef __attribute__((ext_vector_type(4)))  unsigned short u16x4;
typedef __attribute__((ext_vector_type(4)))  unsigned int   tdm_g0_t;
typedef __attribute__((ext_vector_type(8)))  int            tdm_g1_t;
typedef __attribute__((ext_vector_type(4)))  int            tdm_g2_t;

union FragU { u32x4 q[2]; v16bf bf; };

__device__ __forceinline__ unsigned short f2bf(float f) {
  unsigned u = __float_as_uint(f);
  u += 0x7fffu + ((u >> 16) & 1u);      // round-to-nearest-even
  return (unsigned short)(u >> 16);
}

// A/B fragment (16x32 bf16): lane holds row (lane&15); khalf=lane>>4 selects
// k = khalf*8..+7 (vec elems 0..7) and k = 16+khalf*8..+7 (vec elems 8..15).
__device__ __forceinline__ v16bf ldfrag(const unsigned short* p) {
  FragU f;
  f.q[0] = *reinterpret_cast<const u32x4*>(p);
  f.q[1] = *reinterpret_cast<const u32x4*>(p + 16);
  return f.bf;
}

// ---------------------------------------------------------------------------
// TDM: load a 32(k-contig) x 128(rows) bf16 tile into LDS, with the TDM's own
// LDS padding (16 DWORDs data -> +4 DWORDs pad) => 80B LDS row stride, which
// makes 16-lane ds_load_b128 fragment reads bank-conflict-free (20L mod 64).
// D# bitfields per CDNA5 ISA ch.8 (group0 128b, group1 256b, group2/3 zero).
// ---------------------------------------------------------------------------
#define LDSROW 40   // ushorts per LDS tile row (64B data + 16B TDM pad)

__device__ __forceinline__ void tdm_load_tile(unsigned ldsOff,
                                              const unsigned short* gsrc,
                                              unsigned tensorRows) {
  const unsigned long long ga = (unsigned long long)gsrc;
  tdm_g0_t g0;
  g0[0] = 1u;                                          // count=1 (valid), user mode
  g0[1] = ldsOff;                                      // lds_addr (bytes)
  g0[2] = (unsigned)(ga & 0xffffffffull);              // global_addr[31:0]
  g0[3] = (unsigned)((ga >> 32) & 0x1ffffffull)        // global_addr[56:32]
          | (2u << 30);                                // type=2 ("image")
  tdm_g1_t g1;
  g1[0] = (int)((1u << 16)        // data_size = 1 -> 2 bytes (bf16)
              | (1u << 20)        // pad_enable
              | (3u << 22)        // pad_interval: 2^(3+1)=16 DWORDs (one 64B row)
              | (3u << 25));      // pad_amount: 3+1 = 4 DWORDs (16B)
  g1[1] = (int)(1024u << 16);                          // tensor_dim0 = 1024 (k extent)
  g1[2] = (int)((tensorRows & 0xffffu) << 16);         // tensor_dim0 hi=0 | dim1 lo
  g1[3] = (int)((tensorRows >> 16) | (32u << 16));     // dim1 hi | tile_dim0 = 32
  g1[4] = (int)128u;                                   // tile_dim1 = 128, tile_dim2 = 0
  g1[5] = (int)1024;                                   // tensor_dim0_stride = 1024 (lo)
  g1[6] = 0;                                           // stride0 hi | stride1 lo
  g1[7] = 0;
  const tdm_g2_t gz = {0, 0, 0, 0};
#if __has_include(<hip/amd_detail/amd_gfx1250_TDM.h>)
  const tdm_g1_t gz8 = {0, 0, 0, 0, 0, 0, 0, 0};
  __builtin_amdgcn_tensor_load_to_lds(g0, g1, gz, gz, gz8, 0);
#else
  __builtin_amdgcn_tensor_load_to_lds(g0, g1, gz, gz, 0);
#endif
}

// ---------------------------------------------------------------------------
// fp32 -> bf16 conversion (vectorized x4)
// ---------------------------------------------------------------------------
__global__ void __launch_bounds__(256)
cvt_bf16(const float* __restrict__ in, unsigned short* __restrict__ out, long n4) {
  long i = (long)blockIdx.x * blockDim.x + threadIdx.x;
  if (i < n4) {
    f32x4 v = reinterpret_cast<const f32x4*>(in)[i];
    u16x4 o;
    o[0] = f2bf(v[0]); o[1] = f2bf(v[1]); o[2] = f2bf(v[2]); o[3] = f2bf(v[3]);
    reinterpret_cast<u16x4*>(out)[i] = o;
  }
}

// ---------------------------------------------------------------------------
// Generic GEMM: Out[m,n] = sum_k A[m,k] * Bw[n,k] + bias[n]   (A @ Bw^T)
// Block tile 128x128, 8 waves (4 along M x 2 along N), wave tile 32x64.
// TDM double-buffers 128x32 A and B tiles into LDS (wave 0 issues + waits
// s_wait_tensorcnt; workgroup barrier publishes the buffer).
// MODE bit0: fused rotary epilogue. MODE bit1: fp32 output (else bf16).
// ---------------------------------------------------------------------------
template <int MODE>
__global__ void __launch_bounds__(256)
gemm_wmma(const unsigned short* __restrict__ A,
          const unsigned short* __restrict__ Bw,
          const float* __restrict__ bias,
          void* __restrict__ Out) {
  __shared__ unsigned short ldsA[2][128 * LDSROW];
  __shared__ unsigned short ldsB[2][128 * LDSROW];

  const int lane  = threadIdx.x & 31;
  const int wid   = threadIdx.x >> 5;
  const int waveM = wid & 3;
  const int waveN = wid >> 2;
  const int mBlk  = blockIdx.y * 128;
  const int nBlk  = blockIdx.x * 128;
  const int m0    = mBlk + waveM * 32;
  const int n0    = nBlk + waveN * 64;
  const int r15   = lane & 15;
  const int khalf = lane >> 4;

  __builtin_prefetch(bias + n0, 0, 0);   // global_prefetch_b8

  v8f acc[2][4];
#pragma unroll
  for (int i = 0; i < 2; ++i)
#pragma unroll
    for (int j = 0; j < 4; ++j)
#pragma unroll
      for (int r = 0; r < 8; ++r) acc[i][j][r] = 0.0f;

  const unsigned short* gA = A + (size_t)mBlk * WDIM;
  const unsigned short* gB = Bw + (size_t)nBlk * WDIM;

  if (wid == 0) {
    tdm_load_tile((unsigned)(unsigned long long)&ldsA[0][0], gA, (unsigned)NTOK);
    tdm_load_tile((unsigned)(unsigned long long)&ldsB[0][0], gB, (unsigned)WDIM);
    __builtin_amdgcn_s_wait_tensorcnt(0);
  }

  int buf = 0;
  for (int k = 0; k < WDIM; k += 32, buf ^= 1) {
    __syncthreads();   // publish buf; prior reads of buf^1 are done
    if (wid == 0 && k + 32 < WDIM) {
      tdm_load_tile((unsigned)(unsigned long long)&ldsA[buf ^ 1][0], gA + k + 32,
                    (unsigned)NTOK);
      tdm_load_tile((unsigned)(unsigned long long)&ldsB[buf ^ 1][0], gB + k + 32,
                    (unsigned)WDIM);
    }
    const unsigned short* la = &ldsA[buf][(waveM * 32 + r15) * LDSROW + khalf * 8];
    const unsigned short* lb = &ldsB[buf][(waveN * 64 + r15) * LDSROW + khalf * 8];
    v16bf a0 = ldfrag(la);
    v16bf a1 = ldfrag(la + 16 * LDSROW);
#pragma unroll
    for (int j = 0; j < 4; ++j) {
      v16bf bj = ldfrag(lb + j * 16 * LDSROW);
      acc[0][j] = __builtin_amdgcn_wmma_f32_16x16x32_bf16(
          false, a0, false, bj, (short)0, acc[0][j], false, false);
      acc[1][j] = __builtin_amdgcn_wmma_f32_16x16x32_bf16(
          false, a1, false, bj, (short)0, acc[1][j], false, false);
    }
    if (wid == 0 && k + 32 < WDIM) __builtin_amdgcn_s_wait_tensorcnt(0);
  }

  // Epilogue: bias (+ rotary), store. C layout: lane -> n = lane&15,
  // VGPR r -> m_local = r + 8*(lane>>4).
#pragma unroll
  for (int i = 0; i < 2; ++i) {
#pragma unroll
    for (int j = 0; j < 4; ++j) {
      const int n    = n0 + j * 16 + r15;
      const float bn = bias[n];
#pragma unroll
      for (int r = 0; r < 8; ++r) {
        const int m = m0 + i * 16 + r + 8 * khalf;
        float val = acc[i][j][r] + bn;
        if (MODE & 1) {
          // rotary: pairs (2t, 2t+1) are adjacent channels -> adjacent lanes.
          const int s = m >> 8;  // row -> sequence index (m = s*256 + b)
          const float ang =
              (float)s * __expf((float)(n & ~1) * (-9.210340371976184f / (float)WDIM));
          float c, sn;
          __sincosf(ang, &sn, &c);
          const float partner = __shfl_xor(val, 1, 32);
          val = (n & 1) ? (partner * sn + val * c) : (val * c - partner * sn);
        }
        if (MODE & 2)
          ((float*)Out)[(size_t)m * WDIM + n] = val;
        else
          ((unsigned short*)Out)[(size_t)m * WDIM + n] = f2bf(val);
      }
    }
  }
}

// ---------------------------------------------------------------------------
// Attention: one block per (q-tile of 16 rows, batch*head). 4 waves.
// scores = (qh @ kh^T) * scale + mask ; softmax ; o = attn @ vh
// S padded 97 -> 128 (padded K columns get -inf -> attn == 0).
// ---------------------------------------------------------------------------
#define VT_STRIDE 136   // ushorts; 272B rows -> 16B aligned, conflict-free
#define SC_STRIDE 132   // floats
#define AT_STRIDE 136   // ushorts

__global__ void __launch_bounds__(128)
attn_wmma(const unsigned short* __restrict__ Q,
          const unsigned short* __restrict__ K,
          const unsigned short* __restrict__ V,
          const float* __restrict__ mask,
          unsigned short* __restrict__ O) {
  __shared__ unsigned short vT[64 * VT_STRIDE];   // V^T: vT[d][sk]
  __shared__ float          sc[16 * SC_STRIDE];   // raw scores
  __shared__ unsigned short at[16 * AT_STRIDE];   // attn weights (bf16)

  const int tid   = threadIdx.x;
  const int lane  = tid & 31;
  const int w     = tid >> 5;        // wave 0..3
  const int qt    = blockIdx.x;      // q-tile 0..6
  const int bh    = blockIdx.y;      // 0..4095
  const int b     = bh >> 4;
  const int h     = bh & 15;
  const int r15   = lane & 15;
  const int khalf = lane >> 4;
  const size_t headOff = (size_t)h * HDIM;

  auto rowPtr = [&](const unsigned short* P, int s) {
    return P + ((size_t)s * BATCH + b) * WDIM + headOff;
  };

  // ---- stage V transposed into LDS (rows beyond S clamped; they only ever
  //      multiply attn == 0, so finiteness is all that matters) ----
  for (int idx = tid; idx < 128 * 8; idx += 128) {
    const int sk = idx >> 3;
    const int d0 = (idx & 7) * 8;
    const int sks = sk < SLEN ? sk : SLEN - 1;
    const u32x4 vv = *reinterpret_cast<const u32x4*>(rowPtr(V, sks) + d0);
    const unsigned short* pv = reinterpret_cast<const unsigned short*>(&vv);
#pragma unroll
    for (int t = 0; t < 8; ++t) vT[(d0 + t) * VT_STRIDE + sk] = pv[t];
  }

  // ---- scores: each wave owns n-subtiles {w, w+4}; K = HDIM = 2 steps ----
  {
    int mrow = qt * 16 + r15; if (mrow > SLEN - 1) mrow = SLEN - 1;
    const unsigned short* qp = rowPtr(Q, mrow) + khalf * 8;
#pragma unroll
    for (int nt2 = 0; nt2 < 2; ++nt2) {
      const int nt = w + nt2 * 4;
      int srow = nt * 16 + r15; if (srow > SLEN - 1) srow = SLEN - 1;
      const unsigned short* kp = rowPtr(K, srow) + khalf * 8;
      v8f sacc;
#pragma unroll
      for (int r = 0; r < 8; ++r) sacc[r] = 0.0f;
#pragma unroll
      for (int kk = 0; kk < 2; ++kk) {
        v16bf af = ldfrag(qp + kk * 32);
        v16bf bf = ldfrag(kp + kk * 32);
        sacc = __builtin_amdgcn_wmma_f32_16x16x32_bf16(
            false, af, false, bf, (short)0, sacc, false, false);
      }
#pragma unroll
      for (int r = 0; r < 8; ++r)
        sc[(r + 8 * khalf) * SC_STRIDE + nt * 16 + r15] = sacc[r];
    }
  }
  __syncthreads();

  // ---- softmax: 8 threads per row, 128 cols; cols >= S get -inf ----
  {
    const float scale = 0.125f;  // 1/sqrt(64)
    const int row = tid >> 3;
    const int sub = tid & 7;
    int sq = qt * 16 + row; if (sq > SLEN - 1) sq = SLEN - 1;
    float vals[16];
    float mx = -3.0e38f;
#pragma unroll
    for (int i = 0; i < 16; ++i) {
      const int c = sub + i * 8;
      float v = (c < SLEN) ? (sc[row * SC_STRIDE + c] * scale + mask[sq * SLEN + c])
                           : -3.0e38f;
      vals[i] = v;
      mx = fmaxf(mx, v);
    }
    mx = fmaxf(mx, __shfl_xor(mx, 1, 8));
    mx = fmaxf(mx, __shfl_xor(mx, 2, 8));
    mx = fmaxf(mx, __shfl_xor(mx, 4, 8));
    float sum = 0.0f;
#pragma unroll
    for (int i = 0; i < 16; ++i) {
      const float e = __expf(vals[i] - mx);
      vals[i] = e;
      sum += e;
    }
    sum += __shfl_xor(sum, 1, 8);
    sum += __shfl_xor(sum, 2, 8);
    sum += __shfl_xor(sum, 4, 8);
    const float inv = 1.0f / sum;
#pragma unroll
    for (int i = 0; i < 16; ++i)
      at[row * AT_STRIDE + sub + i * 8] = f2bf(vals[i] * inv);
  }
  __syncthreads();

  // ---- o = attn @ vh : wave w owns d-subtile w; K = 128 -> 4 steps ----
  {
    v8f acc;
#pragma unroll
    for (int r = 0; r < 8; ++r) acc[r] = 0.0f;
    const unsigned short* ap = &at[r15 * AT_STRIDE + khalf * 8];
    const unsigned short* bp = &vT[(w * 16 + r15) * VT_STRIDE + khalf * 8];
#pragma unroll
    for (int kt = 0; kt < 4; ++kt) {
      v16bf af = ldfrag(ap + kt * 32);
      v16bf bf = ldfrag(bp + kt * 32);
      acc = __builtin_amdgcn_wmma_f32_16x16x32_bf16(
          false, af, false, bf, (short)0, acc, false, false);
    }
#pragma unroll
    for (int r = 0; r < 8; ++r) {
      const int m  = r + 8 * khalf;
      const int sq = qt * 16 + m;
      if (sq < SLEN) {
        const int d = w * 16 + r15;
        O[((size_t)sq * BATCH + b) * WDIM + headOff + d] = f2bf(acc[r]);
      }
    }
  }
}

// ---------------------------------------------------------------------------
// Host-side orchestration
// ---------------------------------------------------------------------------
extern "C" void kernel_launch(void* const* d_in, const int* in_sizes, int n_in,
                              void* d_out, int out_size, void* d_ws, size_t ws_size,
                              hipStream_t stream) {
  const float* tensor    = (const float*)d_in[0];
  const float* mask      = (const float*)d_in[1];
  const float* q_w       = (const float*)d_in[2];
  const float* q_b       = (const float*)d_in[3];
  const float* k_w       = (const float*)d_in[4];
  const float* k_b       = (const float*)d_in[5];
  const float* v_w       = (const float*)d_in[6];
  const float* v_b       = (const float*)d_in[7];
  const float* in_proj_w = (const float*)d_in[8];
  const float* in_proj_b = (const float*)d_in[9];
  const float* mha_out_w = (const float*)d_in[10];
  const float* mha_out_b = (const float*)d_in[11];
  const float* out_w     = (const float*)d_in[12];
  const float* out_b     = (const float*)d_in[13];

  unsigned short* ws = (unsigned short*)d_ws;
  unsigned short* t_bf  = ws;
  unsigned short* q1    = ws + 1 * TOKW;
  unsigned short* k1    = ws + 2 * TOKW;
  unsigned short* v1    = ws + 3 * TOKW;
  unsigned short* q2    = ws + 4 * TOKW;
  unsigned short* k2    = ws + 5 * TOKW;
  unsigned short* v2    = ws + 6 * TOKW;
  unsigned short* wbase = ws + 7 * TOKW;
  unsigned short* qwb   = wbase;
  unsigned short* kwb   = wbase + 1 * WW;
  unsigned short* vwb   = wbase + 2 * WW;
  unsigned short* ipwb  = wbase + 3 * WW;   // 3*WW elements
  unsigned short* mhawb = wbase + 6 * WW;
  unsigned short* outwb = wbase + 7 * WW;
  unsigned short* o_bf  = q1;               // q1 dead after second projection
  unsigned short* t1    = k1;               // k1 dead after attention

  // fp32 -> bf16 conversions
  {
    long n4 = (long)(TOKW / 4);
    cvt_bf16<<<dim3((unsigned)((n4 + 255) / 256)), 256, 0, stream>>>(tensor, t_bf, n4);
    long w4 = (long)(WW / 4);
    unsigned wb = (unsigned)((w4 + 255) / 256);
    cvt_bf16<<<dim3(wb), 256, 0, stream>>>(q_w, qwb, w4);
    cvt_bf16<<<dim3(wb), 256, 0, stream>>>(k_w, kwb, w4);
    cvt_bf16<<<dim3(wb), 256, 0, stream>>>(v_w, vwb, w4);
    cvt_bf16<<<dim3(3 * wb), 256, 0, stream>>>(in_proj_w, ipwb, 3 * w4);
    cvt_bf16<<<dim3(wb), 256, 0, stream>>>(mha_out_w, mhawb, w4);
    cvt_bf16<<<dim3(wb), 256, 0, stream>>>(out_w, outwb, w4);
  }

  const dim3 ggrid(WDIM / 128, NTOK / 128);   // (8, 194)
  const dim3 gblk(256);

  // QKV projection (+rotary on q,k)
  gemm_wmma<1><<<ggrid, gblk, 0, stream>>>(t_bf, qwb, q_b, q1);
  gemm_wmma<1><<<ggrid, gblk, 0, stream>>>(t_bf, kwb, k_b, k1);
  gemm_wmma<0><<<ggrid, gblk, 0, stream>>>(t_bf, vwb, v_b, v1);

  // in_proj q/k/v
  gemm_wmma<0><<<ggrid, gblk, 0, stream>>>(q1, ipwb,          in_proj_b,        q2);
  gemm_wmma<0><<<ggrid, gblk, 0, stream>>>(k1, ipwb + 1 * WW, in_proj_b + 1024, k2);
  gemm_wmma<0><<<ggrid, gblk, 0, stream>>>(v1, ipwb + 2 * WW, in_proj_b + 2048, v2);

  // attention
  attn_wmma<<<dim3(7, BATCH * NHEAD), dim3(128), 0, stream>>>(q2, k2, v2, mask, o_bf);

  // output projections (final one in fp32 straight to d_out)
  gemm_wmma<0><<<ggrid, gblk, 0, stream>>>(o_bf, mhawb, mha_out_b, t1);
  gemm_wmma<2><<<ggrid, gblk, 0, stream>>>(t1, outwb, out_b, (float*)d_out);
}
